// NodePairTrainableKernel_30288109371691
// MI455X (gfx1250) — matrix operand order repaired and checked by reading, not compiled, not run on hardware
//
#include <hip/hip_runtime.h>

// ---------------------------------------------------------------------------
// NodePair MLP on MI455X (gfx1250).
//
// Roofline: output is 2*32*128*128*32 f32 = 134 MB -> ~5.8us at 23.3 TB/s.
// Pair GEMMs are 2 x (1M x 32 x 32) = 4.3 GFLOP; using v_wmma_f32_16x16x32_f16
// (16384 FLOP/instr, f32 accumulate) keeps the kernel bandwidth-bound.
// Stage 1 (node MLP, ~46 MFLOP) is negligible; pa/pb (2 MB) live in L2.
// ---------------------------------------------------------------------------

typedef __attribute__((ext_vector_type(16))) _Float16 v16h;
typedef __attribute__((ext_vector_type(8)))  _Float16 v8h;
typedef __attribute__((ext_vector_type(8)))  float    v8f;

#define NBINS    64      // B*NB
#define Pn       128     // particles per bin
#define Fdim     32
#define Hdim     32
#define NOUT     8       // NODE_OUT
#define ODIM     32      // OUT
#define NODES    (NBINS * Pn)   // 8192

__device__ __forceinline__ float elu(float x) {
    return x > 0.0f ? x : (__expf(x) - 1.0f);
}

// ------------------------- Stage 1: node MLP + projections ------------------
__global__ __launch_bounds__(256) void node_mlp_kernel(
    const float* __restrict__ x,   const float* __restrict__ msk,
    const float* __restrict__ w0,  const float* __restrict__ b0,
    const float* __restrict__ w1,  const float* __restrict__ b1,
    const float* __restrict__ w2,  const float* __restrict__ b2,
    const float* __restrict__ pw0,
    float* __restrict__ pa, float* __restrict__ pb)
{
    __shared__ float sW0[Fdim * Hdim], sW1[Hdim * Hdim], sW2[Hdim * NOUT];
    __shared__ float sB0[Hdim], sB1[Hdim], sB2[NOUT];
    __shared__ float sWa[NOUT * Hdim], sWb[NOUT * Hdim];

    const int tid = threadIdx.x;
    for (int i = tid; i < Fdim * Hdim; i += 256) sW0[i] = w0[i];
    for (int i = tid; i < Hdim * Hdim; i += 256) sW1[i] = w1[i];
    for (int i = tid; i < Hdim * NOUT; i += 256) sW2[i] = w2[i];
    for (int i = tid; i < Hdim; i += 256) { sB0[i] = b0[i]; sB1[i] = b1[i]; }
    for (int i = tid; i < NOUT; i += 256) sB2[i] = b2[i];
    // pair_w0 is (16,32): rows 0..7 -> wa, rows 8..15 -> wb
    for (int i = tid; i < NOUT * Hdim; i += 256) {
        sWa[i] = pw0[i];
        sWb[i] = pw0[NOUT * Hdim + i];
    }
    __syncthreads();

    const int node = blockIdx.x * 256 + tid;

    float xi[Fdim];
    #pragma unroll
    for (int k = 0; k < Fdim; ++k) xi[k] = x[(size_t)node * Fdim + k];

    float h0[Hdim];
    #pragma unroll
    for (int j = 0; j < Hdim; ++j) {
        float acc = sB0[j];
        #pragma unroll
        for (int k = 0; k < Fdim; ++k) acc += xi[k] * sW0[k * Hdim + j];
        h0[j] = elu(acc);
    }
    float h1[Hdim];
    #pragma unroll
    for (int j = 0; j < Hdim; ++j) {
        float acc = sB1[j];
        #pragma unroll
        for (int k = 0; k < Hdim; ++k) acc += h0[k] * sW1[k * Hdim + j];
        h1[j] = elu(acc);
    }
    const float m = msk[node];
    float np[NOUT];
    #pragma unroll
    for (int j = 0; j < NOUT; ++j) {
        float acc = sB2[j];
        #pragma unroll
        for (int k = 0; k < Hdim; ++k) acc += h1[k] * sW2[k * NOUT + j];
        np[j] = elu(acc) * m;
    }
    #pragma unroll
    for (int j = 0; j < Hdim; ++j) {
        float pav = 0.0f, pbv = 0.0f;
        #pragma unroll
        for (int k = 0; k < NOUT; ++k) {
            pav += np[k] * sWa[k * Hdim + j];
            pbv += np[k] * sWb[k * Hdim + j];
        }
        pa[(size_t)node * Hdim + j] = pav;
        pb[(size_t)node * Hdim + j] = pbv;
    }
}

// ------------------------- Stage 2: pair MLP via WMMA -----------------------
// Grid: 512 blocks = 64 bins x 8 i-tiles. Block = 256 thr = 8 waves.
// Wave w handles j in [16w, 16w+16); loops over 16 i rows; per i:
//   A (16 pairs x K=32, f16)  ->  2x wmma (W1)  ->  elu -> 2x wmma (W2) -> elu -> out
__global__ __launch_bounds__(256) void pair_mlp_kernel(
    const float* __restrict__ pa,  const float* __restrict__ pb,
    const float* __restrict__ pb0, const float* __restrict__ pw1,
    const float* __restrict__ pb1, const float* __restrict__ pw2,
    const float* __restrict__ pb2, float* __restrict__ out)
{
    __shared__ __align__(16) float    sPa[16 * Hdim];
    __shared__ __align__(16) float    sPb[Pn * Hdim];
    __shared__ __align__(16) _Float16 sW1t[Hdim * Hdim];   // transposed [N][K]
    __shared__ __align__(16) _Float16 sW2t[Hdim * ODIM];   // transposed [N][K]
    __shared__ float sB0[Hdim], sB1[Hdim], sB2[ODIM];
    __shared__ __align__(16) _Float16 stg[8][16 * Hdim];   // per-wave staging

    const int tid = threadIdx.x;
    const int bin = blockIdx.x >> 3;
    const int i0  = (blockIdx.x & 7) * 16;

    for (int idx = tid; idx < 16 * Hdim; idx += 256) {
        int i = idx >> 5, k = idx & 31;
        sPa[idx] = pa[((size_t)(bin * Pn + i0 + i)) * Hdim + k];
    }
    for (int idx = tid; idx < Pn * Hdim; idx += 256)
        sPb[idx] = pb[(size_t)bin * Pn * Hdim + idx];
    for (int idx = tid; idx < Hdim * Hdim; idx += 256) {
        int k = idx >> 5, n = idx & 31;               // pw1[k][n] row-major
        sW1t[n * Hdim + k] = (_Float16)pw1[idx];
        sW2t[n * Hdim + k] = (_Float16)pw2[idx];
    }
    for (int idx = tid; idx < Hdim; idx += 256) {
        sB0[idx] = pb0[idx]; sB1[idx] = pb1[idx]; sB2[idx] = pb2[idx];
    }
    __syncthreads();

    const int lane = tid & 31;
    const int wv   = tid >> 5;
    const int half = lane >> 4;      // 0: lanes 0-15, 1: lanes 16-31
    const int ln   = lane & 15;
    const int j0   = wv * 16;
    _Float16* stgw = &stg[wv][0];

    // --- B fragments (16-bit B, K=32): lanes 0-15 hold K=0..15, lanes 16-31
    //     hold K=16..31; N = lane%16; packed pairs (2v,2v+1) -> contiguous K.
    v16h bw1[2], bw2[2];
    #pragma unroll
    for (int nt = 0; nt < 2; ++nt) {
        const int N  = nt * 16 + ln;
        const int kb = half * 16;
        *(v8h*)&bw1[nt]       = *(const v8h*)&sW1t[N * Hdim + kb];
        *((v8h*)&bw1[nt] + 1) = *(const v8h*)&sW1t[N * Hdim + kb + 8];
        *(v8h*)&bw2[nt]       = *(const v8h*)&sW2t[N * Hdim + kb];
        *((v8h*)&bw2[nt] + 1) = *(const v8h*)&sW2t[N * Hdim + kb + 8];
    }
    // --- bias preloaded into C accumulator (C/D layout: N = lane%16)
    v8f c1[2], c2[2];
    #pragma unroll
    for (int nt = 0; nt < 2; ++nt) {
        const float v1 = sB1[nt * 16 + ln];
        const float v2 = sB2[nt * 16 + ln];
        #pragma unroll
        for (int r = 0; r < 8; ++r) { c1[nt][r] = v1; c2[nt][r] = v2; }
    }

    // A-tile build mapping: lane -> (row m, k-half)
    const int mrow = lane >> 1;           // 0..15 (pair row within tile)
    const int kcol = (lane & 1) * 16;     // 0 or 16
    const int ab   = half * 16;           // A-fragment K base (contiguous K = 16*half + s)

    for (int i = 0; i < 16; ++i) {
        // ---- build A1 = elu(pa_i + pb_j + b0), f16, into staging [M][K]
        v8h lo, hi;
        #pragma unroll
        for (int t = 0; t < 8; ++t) {
            lo[t] = (_Float16)elu(sPa[i * Hdim + kcol + t] +
                                  sPb[(j0 + mrow) * Hdim + kcol + t] +
                                  sB0[kcol + t]);
            hi[t] = (_Float16)elu(sPa[i * Hdim + kcol + 8 + t] +
                                  sPb[(j0 + mrow) * Hdim + kcol + 8 + t] +
                                  sB0[kcol + 8 + t]);
        }
        *(v8h*)&stgw[mrow * Hdim + kcol]     = lo;
        *(v8h*)&stgw[mrow * Hdim + kcol + 8] = hi;
        __asm__ volatile("s_wait_dscnt 0" ::: "memory");

        // ---- A fragment load (contiguous K order, same as B):
        // lanes 0-15: K=0..15 ; lanes 16-31: K=16..31 ; M=ln
        v16h a;
        *(v8h*)&a       = *(const v8h*)&stgw[ln * Hdim + ab];
        *((v8h*)&a + 1) = *(const v8h*)&stgw[ln * Hdim + ab + 8];

        // ---- layer 1: t = A @ W1 + b1
        v8f d0 = __builtin_amdgcn_wmma_f32_16x16x32_f16(false, a, false, bw1[0],
                                                        (short)0, c1[0], false, false);
        v8f d1 = __builtin_amdgcn_wmma_f32_16x16x32_f16(false, a, false, bw1[1],
                                                        (short)0, c1[1], false, false);

        // ---- elu + f16, scatter back to staging in [M][K] layout
        #pragma unroll
        for (int r = 0; r < 8; ++r) {
            const int m = r + half * 8;          // D layout: M = r + 8*half
            stgw[m * Hdim + ln]      = (_Float16)elu(d0[r]);
            stgw[m * Hdim + 16 + ln] = (_Float16)elu(d1[r]);
        }
        __asm__ volatile("s_wait_dscnt 0" ::: "memory");

        v16h a2;
        *(v8h*)&a2       = *(const v8h*)&stgw[ln * Hdim + ab];
        *((v8h*)&a2 + 1) = *(const v8h*)&stgw[ln * Hdim + ab + 8];

        // ---- layer 2: dm = t @ W2 + b2
        v8f e0 = __builtin_amdgcn_wmma_f32_16x16x32_f16(false, a2, false, bw2[0],
                                                        (short)0, c2[0], false, false);
        v8f e1 = __builtin_amdgcn_wmma_f32_16x16x32_f16(false, a2, false, bw2[1],
                                                        (short)0, c2[1], false, false);

        // ---- final elu + store: out[bin][i0+i][j0+m][n]
        float* orow = out + ((size_t)(bin * Pn + i0 + i) * Pn + j0) * ODIM;
        #pragma unroll
        for (int r = 0; r < 8; ++r) {
            const int m = r + half * 8;
            orow[m * ODIM + ln]      = elu(e0[r]);
            orow[m * ODIM + 16 + ln] = elu(e1[r]);
        }
    }
}

// ---------------------------------------------------------------------------
extern "C" void kernel_launch(void* const* d_in, const int* in_sizes, int n_in,
                              void* d_out, int out_size, void* d_ws, size_t ws_size,
                              hipStream_t stream) {
    (void)in_sizes; (void)n_in; (void)out_size; (void)ws_size;

    const float* x    = (const float*)d_in[0];
    const float* msk  = (const float*)d_in[1];
    const float* nw0  = (const float*)d_in[2];
    const float* nb0  = (const float*)d_in[3];
    const float* nw1  = (const float*)d_in[4];
    const float* nb1  = (const float*)d_in[5];
    const float* nw2  = (const float*)d_in[6];
    const float* nb2  = (const float*)d_in[7];
    const float* pw0  = (const float*)d_in[8];
    const float* pb0  = (const float*)d_in[9];
    const float* pw1  = (const float*)d_in[10];
    const float* pb1  = (const float*)d_in[11];
    const float* pw2  = (const float*)d_in[12];
    const float* pb2  = (const float*)d_in[13];
    float* out = (float*)d_out;

    float* pa = (float*)d_ws;                     // 8192*32 f32 = 1 MB
    float* pb = pa + (size_t)NODES * Hdim;        // 8192*32 f32 = 1 MB

    node_mlp_kernel<<<NODES / 256, 256, 0, stream>>>(
        x, msk, nw0, nb0, nw1, nb1, nw2, nb2, pw0, pa, pb);

    pair_mlp_kernel<<<NBINS * 8, 256, 0, stream>>>(
        pa, pb, pb0, pw1, pb1, pw2, pb2, out);
}
